// TemporalChannelInteractionFusionModule_67920612819381
// MI455X (gfx1250) — matrix operand order repaired and checked by reading, not compiled, not run on hardware
//
#include <hip/hip_runtime.h>
#include <hip/hip_bf16.h>

// ---------------------------------------------------------------------------
// Problem constants: T=64, N=4096, F=128, NHID=128, RNN=128
// ---------------------------------------------------------------------------
constexpr int T    = 64;
constexpr int N    = 4096;
constexpr int F    = 128;
constexpr int NHID = 128;
constexpr int RNN  = 128;
constexpr int NF   = N * F;     // 524288
constexpr int FC   = F + NHID;  // 256 (GRU input width)
constexpr int H3   = 3 * RNN;   // 384

typedef __attribute__((ext_vector_type(16))) _Float16 v16h;
typedef __attribute__((ext_vector_type(8)))  _Float16 h8;
typedef __attribute__((ext_vector_type(8)))  float    v8f;
typedef __attribute__((ext_vector_type(4)))  int      i4;

#if __has_builtin(__builtin_amdgcn_global_load_async_to_lds_b128) && \
    __has_builtin(__builtin_amdgcn_s_wait_asynccnt)
#define HAVE_ASYNC_LDS 1
#else
#define HAVE_ASYNC_LDS 0
#endif

__device__ __forceinline__ float fast_rcp(float x) {
#if __has_builtin(__builtin_amdgcn_rcpf)
  return __builtin_amdgcn_rcpf(x);
#else
  return 1.0f / x;
#endif
}
__device__ __forceinline__ float fast_tanh(float x) {
#if __has_builtin(__builtin_amdgcn_tanhf)
  return __builtin_amdgcn_tanhf(x);
#else
  return tanhf(x);
#endif
}
__device__ __forceinline__ float sigmoidf(float x) { return fast_rcp(1.0f + __expf(-x)); }

// ---------------------------------------------------------------------------
// f16 fragment loaders (wave32, v_wmma_f32_16x16x32_f16).
// A (16M x 32K), row-major source: lane m=lane&15, kb=(lane&16)?8:0;
//   halves 0..7 = K kb..kb+7, halves 8..15 = K 16+kb.. -> 2x b128 loads.
// B (32K x 16N) from n-major source (element (k,n) at base[n*ld+k]):
//   same K order as A: halves 0..7 = K kb..kb+7, 8..15 = K 16+kb.. -> 2x b128 loads.
// C/D: lane n=lane&15; VGPR r -> M = r + ((lane&16)?8:0)
// ---------------------------------------------------------------------------
__device__ __forceinline__ v16h load_a16(const _Float16* base, int ld) {
  int lane = threadIdx.x & 31;
  int m  = lane & 15;
  int kb = (lane & 16) ? 8 : 0;
  const _Float16* p = base + m * ld + kb;
  h8 lo = *(const h8*)(p);
  h8 hi = *(const h8*)(p + 16);
  return __builtin_shufflevector(lo, hi, 0, 1, 2, 3, 4, 5, 6, 7,
                                 8, 9, 10, 11, 12, 13, 14, 15);
}
__device__ __forceinline__ v16h load_bk16(const _Float16* base, int ld) {
  int lane = threadIdx.x & 31;
  int n  = lane & 15;
  int kb = (lane & 16) ? 8 : 0;
  const _Float16* p = base + n * ld + kb;
  h8 lo = *(const h8*)(p);
  h8 hi = *(const h8*)(p + 16);
  return __builtin_shufflevector(lo, hi, 0, 1, 2, 3, 4, 5, 6, 7,
                                 8, 9, 10, 11, 12, 13, 14, 15);
}

#define WMMA_F16(A, B, C) \
  __builtin_amdgcn_wmma_f32_16x16x32_f16(false, (A), false, (B), (short)0, (C), false, false)

// ---------------------------------------------------------------------------
// K1: time-normalization -> f16 norm_x in (t,c) layout AND transposed (c,t)
// ---------------------------------------------------------------------------
__global__ void norm_kernel(const float* __restrict__ x,
                            _Float16* __restrict__ nx, _Float16* __restrict__ nxT) {
  int c = blockIdx.x * blockDim.x + threadIdx.x;
  if (c >= NF) return;
  float s = 0.f, ss = 0.f;
#pragma unroll 4
  for (int t = 0; t < T; ++t) {
    float v = x[(size_t)t * NF + c];
    s += v; ss += v * v;
  }
  float m   = s * (1.0f / T);
  float var = fmaxf((ss - (float)T * m * m) * (1.0f / (T - 1)), 0.0f);
  float r   = 1.0f / (sqrtf(var) + 1e-6f);
#pragma unroll 4
  for (int t = 0; t < T; ++t) {
    _Float16 h = (_Float16)((x[(size_t)t * NF + c] - m) * r);
    nx[(size_t)t * NF + c] = h;
    nxT[(size_t)c * T + t] = h;
  }
}

// ---------------------------------------------------------------------------
// K2: plain f32 -> f16 conversion
// ---------------------------------------------------------------------------
__global__ void cvt16_kernel(const float* __restrict__ src, _Float16* __restrict__ dst,
                             size_t n) {
  size_t i = (size_t)blockIdx.x * blockDim.x + threadIdx.x;
  if (i < n) dst[i] = (_Float16)src[i];
}

// ---------------------------------------------------------------------------
// K3: z_t[t,f] = mean_n norm_x[t,n,f]
// ---------------------------------------------------------------------------
__global__ void zt_kernel(const _Float16* __restrict__ nx, float* __restrict__ zt) {
  int t = blockIdx.x;
  int f = threadIdx.x;
  const _Float16* p = nx + (size_t)t * NF;
  float acc = 0.f;
  for (int n = 0; n < N; ++n) acc += (float)p[n * F + f];
  zt[t * F + f] = acc * (1.0f / N);
}

// ---------------------------------------------------------------------------
// K4: gamma_tem / gamma_cha (tiny softmaxes), single 128-thread block
// ---------------------------------------------------------------------------
__global__ void gamma_kernel(const float* __restrict__ zt,
                             const float* __restrict__ W_tem3, const float* __restrict__ b_tem3,
                             const float* __restrict__ W_cha3, const float* __restrict__ b_cha3,
                             float* __restrict__ gtem, float* __restrict__ gcha) {
  int h = threadIdx.x;  // 0..127
  if (h < T) gtem[h] = 0.f;
  gcha[h] = 0.f;
  __syncthreads();

  float lt[T];
  float mx = -1e30f;
  for (int t = 0; t < T; ++t) {
    float s = b_tem3[t];
    for (int f = 0; f < F; ++f) s += W_tem3[h * F + f] * zt[t * F + f];
    lt[t] = s;
    mx = fmaxf(mx, s);
  }
  float se = 0.f;
  for (int t = 0; t < T; ++t) { lt[t] = __expf(lt[t] - mx); se += lt[t]; }
  float inv = fast_rcp(se * (float)NHID);
  for (int t = 0; t < T; ++t) atomicAdd(&gtem[t], lt[t] * inv);

  float lf[F];
  mx = -1e30f;
  for (int f = 0; f < F; ++f) {
    float s = b_cha3[f];
    for (int t = 0; t < T; ++t) s += W_cha3[h * T + t] * zt[t * F + f];
    lf[f] = s;
    mx = fmaxf(mx, s);
  }
  se = 0.f;
  for (int f = 0; f < F; ++f) { lf[f] = __expf(lf[f] - mx); se += lf[f]; }
  inv = fast_rcp(se * (float)NHID);
  for (int f = 0; f < F; ++f) atomicAdd(&gcha[f], lf[f] * inv);
}

// ---------------------------------------------------------------------------
// K5: small dense GEMM folds, f16 output, optional C^T
// ---------------------------------------------------------------------------
__global__ void smallgemm16_kernel(const float* __restrict__ A, const float* __restrict__ B,
                                   _Float16* __restrict__ C, int M, int Nc, int K,
                                   int transpose) {
  int idx = blockIdx.x * blockDim.x + threadIdx.x;
  if (idx >= M * Nc) return;
  int m = idx / Nc, n = idx % Nc;
  float s = 0.f;
  for (int k = 0; k < K; ++k) s += A[m * K + k] * B[k * Nc + n];
  if (transpose) C[n * M + m] = (_Float16)s;
  else           C[m * Nc + n] = (_Float16)s;
}

// ---------------------------------------------------------------------------
// K6: S_tem = sigmoid( A_tem(64x64) @ norm_x(64 x NF) ) * gamma_tem[t] -> f16
// Register-blocked over all 4 t-tiles: one streaming B fragment (norm_x^T)
// feeds 4 WMMAs against cached A_tem tiles. grid (NF/16), 1 wave/block.
// ---------------------------------------------------------------------------
__global__ void stem_kernel(const _Float16* __restrict__ nxT, const _Float16* __restrict__ Atem,
                            const float* __restrict__ gtem, _Float16* __restrict__ sb) {
  int c0 = blockIdx.x * 16;
  v8f acc[4] = {};
  for (int kt = 0; kt < T; kt += 32) {
    v16h b = load_bk16(nxT + (size_t)c0 * T + kt, T);
#pragma unroll
    for (int tt = 0; tt < 4; ++tt) {
      v16h a = load_a16(Atem + (tt * 16) * T + kt, T);
      acc[tt] = WMMA_F16(a, b, acc[tt]);
    }
  }
  int lane = threadIdx.x;
  int n = lane & 15, hi = (lane & 16) ? 8 : 0;
#pragma unroll
  for (int tt = 0; tt < 4; ++tt) {
#pragma unroll
    for (int r = 0; r < 8; ++r) {
      int t = tt * 16 + r + hi;
      sb[(size_t)t * NF + c0 + n] = (_Float16)(sigmoidf(acc[tt][r]) * gtem[t]);
    }
  }
}

// ---------------------------------------------------------------------------
// K7 (fused scha+gate): per wave, a full 16-row x 128-col block.
// Phase 1: cha = norm_x @ B_cha over all 8 col tiles (A frag reused 8x);
//          S_fused = sigmoid(cha)*gamma_cha*S_tem -> 4 KB LDS tile.
// Phase 2: re-read S_fused from LDS as A fragments (ds_load_b128), run the
//          8x2 gate/filt WMMAs, S_int = sigmoid(.)*tanh(.) -> f16 global.
// S_fused never touches HBM (saves ~134 MB round trip).
// grid (T*N/16), 1 wave/block.
// ---------------------------------------------------------------------------
__global__ void fuse_kernel(const _Float16* __restrict__ nx, const _Float16* __restrict__ BchaT,
                            const float* __restrict__ gcha, const _Float16* __restrict__ sb,
                            const _Float16* __restrict__ Wt, const _Float16* __restrict__ Gt,
                            _Float16* __restrict__ si) {
  __shared__ __align__(16) _Float16 sf[16 * F];  // S_fused row-block
  size_t row0 = (size_t)blockIdx.x * 16;         // row = t*N + n
  int lane = threadIdx.x;
  int n = lane & 15, hi = (lane & 16) ? 8 : 0;

  // Phase 1: channel-interaction GEMM, all 8 column tiles.
  v8f acc[8] = {};
  for (int kt = 0; kt < F; kt += 32) {
    v16h a = load_a16(nx + row0 * F + kt, F);
#pragma unroll
    for (int c = 0; c < 8; ++c) {
      v16h b = load_bk16(BchaT + (c * 16) * F + kt, F);
      acc[c] = WMMA_F16(a, b, acc[c]);
    }
  }
#pragma unroll
  for (int c = 0; c < 8; ++c) {
    int f = c * 16 + n;
    float g = gcha[f];
#pragma unroll
    for (int r = 0; r < 8; ++r) {
      int m = r + hi;
      float stem_v = (float)sb[(row0 + m) * F + f];
      sf[m * F + f] = (_Float16)(sigmoidf(acc[c][r]) * g * stem_v);
    }
  }
  __syncthreads();

  // Phase 2: dual-gate GEMMs from the LDS S_fused tile.
  v8f ag[8] = {}, af[8] = {};
  for (int kt = 0; kt < F; kt += 32) {
    v16h a = load_a16(&sf[kt], F);
#pragma unroll
    for (int c = 0; c < 8; ++c) {
      v16h bg = load_bk16(Wt + (c * 16) * F + kt, F);
      v16h bf = load_bk16(Gt + (c * 16) * F + kt, F);
      ag[c] = WMMA_F16(a, bg, ag[c]);
      af[c] = WMMA_F16(a, bf, af[c]);
    }
  }
#pragma unroll
  for (int c = 0; c < 8; ++c) {
#pragma unroll
    for (int r = 0; r < 8; ++r) {
      si[(row0 + r + hi) * NHID + c * 16 + n] =
          (_Float16)(sigmoidf(ag[c][r]) * fast_tanh(af[c][r]));
    }
  }
}

// ---------------------------------------------------------------------------
// K8: zero fills
// ---------------------------------------------------------------------------
__global__ void fill0f_kernel(float* __restrict__ p, int n) {
  int i = blockIdx.x * blockDim.x + threadIdx.x;
  if (i < n) p[i] = 0.f;
}
__global__ void fill0h_kernel(_Float16* __restrict__ p, int n) {
  int i = blockIdx.x * blockDim.x + threadIdx.x;
  if (i < n) p[i] = (_Float16)0.f;
}

// ---------------------------------------------------------------------------
// K9: one GRU step. 256-thread block = 8 waves sharing one hidden-column
// tile j0; W_ih/W_hh slices (36 KB f16) staged in LDS — via async
// GLOBAL_LOAD_ASYNC_TO_LDS_B128 (ASYNCcnt) when the builtin exists, else
// b128 register staging. B fragments come from ds_load_b128.
// grid (N/128, RNN/16), block 256.
// ---------------------------------------------------------------------------
__global__ void gru_step_kernel(int t,
                                const _Float16* __restrict__ x16,
                                const _Float16* __restrict__ si16,
                                const _Float16* __restrict__ Wih, const _Float16* __restrict__ Whh,
                                const float* __restrict__ b_ih, const float* __restrict__ b_hh,
                                const float* __restrict__ h_in, const _Float16* __restrict__ h16_in,
                                float* __restrict__ h_out, _Float16* __restrict__ h16_out) {
  __shared__ __align__(16) _Float16 sWih[48 * FC];   // 3 gates x 16 cols x 256 K
  __shared__ __align__(16) _Float16 sWhh[48 * RNN];  // 3 gates x 16 cols x 128 K

  int tid  = threadIdx.x;
  int lane = tid & 31;
  int w    = tid >> 5;
  int m0 = (blockIdx.x * 8 + w) * 16;
  int j0 = blockIdx.y * 16;

  // Stage the block's weight slices to LDS.
  for (int i = tid; i < 48 * (FC / 8); i += 256) {
    int row = i >> 5;  // FC/8 == 32 chunks per row
    int k   = (i & 31) * 8;
    int grow = j0 + (row >> 4) * RNN + (row & 15);
#if HAVE_ASYNC_LDS
    __builtin_amdgcn_global_load_async_to_lds_b128(
        (__attribute__((address_space(1))) i4*)&Wih[(size_t)grow * FC + k],
        (__attribute__((address_space(3))) i4*)&sWih[row * FC + k], 0, 0);
#else
    *(h8*)&sWih[row * FC + k] = *(const h8*)&Wih[(size_t)grow * FC + k];
#endif
  }
  for (int i = tid; i < 48 * (RNN / 8); i += 256) {
    int row = i >> 4;  // RNN/8 == 16 chunks per row
    int k   = (i & 15) * 8;
    int grow = j0 + (row >> 4) * RNN + (row & 15);
#if HAVE_ASYNC_LDS
    __builtin_amdgcn_global_load_async_to_lds_b128(
        (__attribute__((address_space(1))) i4*)&Whh[(size_t)grow * RNN + k],
        (__attribute__((address_space(3))) i4*)&sWhh[row * RNN + k], 0, 0);
#else
    *(h8*)&sWhh[row * RNN + k] = *(const h8*)&Whh[(size_t)grow * RNN + k];
#endif
  }
#if HAVE_ASYNC_LDS
  __builtin_amdgcn_s_wait_asynccnt(0);
#endif
  __syncthreads();

  const _Float16* xt = x16 + (size_t)t * NF + (size_t)m0 * F;
  const _Float16* st = si16 + (size_t)t * N * NHID + (size_t)m0 * NHID;

  v8f ai0 = {}, ai1 = {}, ai2 = {}, ah0 = {}, ah1 = {}, ah2 = {};

  for (int kt = 0; kt < FC; kt += 32) {
    const _Float16* ab = (kt < F) ? (xt + kt) : (st + (kt - F));
    v16h a  = load_a16(ab, F);
    v16h b0 = load_bk16(&sWih[0 * 16 * FC + kt], FC);
    v16h b1 = load_bk16(&sWih[1 * 16 * FC + kt], FC);
    v16h b2 = load_bk16(&sWih[2 * 16 * FC + kt], FC);
    ai0 = WMMA_F16(a, b0, ai0);
    ai1 = WMMA_F16(a, b1, ai1);
    ai2 = WMMA_F16(a, b2, ai2);
  }
  for (int kt = 0; kt < RNN; kt += 32) {
    v16h a  = load_a16(h16_in + (size_t)m0 * RNN + kt, RNN);
    v16h b0 = load_bk16(&sWhh[0 * 16 * RNN + kt], RNN);
    v16h b1 = load_bk16(&sWhh[1 * 16 * RNN + kt], RNN);
    v16h b2 = load_bk16(&sWhh[2 * 16 * RNN + kt], RNN);
    ah0 = WMMA_F16(a, b0, ah0);
    ah1 = WMMA_F16(a, b1, ah1);
    ah2 = WMMA_F16(a, b2, ah2);
  }

  int n = lane & 15, hi = (lane & 16) ? 8 : 0;
  int j = j0 + n;
  float bi0 = b_ih[j], bi1 = b_ih[j + RNN], bi2 = b_ih[j + 2 * RNN];
  float bh0 = b_hh[j], bh1 = b_hh[j + RNN], bh2 = b_hh[j + 2 * RNN];
#pragma unroll
  for (int r = 0; r < 8; ++r) {
    size_t row = (size_t)(m0 + r + hi);
    float rr = sigmoidf(ai0[r] + bi0 + ah0[r] + bh0);
    float zz = sigmoidf(ai1[r] + bi1 + ah1[r] + bh1);
    float nn = fast_tanh(ai2[r] + bi2 + rr * (ah2[r] + bh2));
    float hp = h_in[row * RNN + j];
    float hv = (1.0f - zz) * nn + zz * hp;
    h_out[row * RNN + j]   = hv;
    h16_out[row * RNN + j] = (_Float16)hv;
  }
}

// ---------------------------------------------------------------------------
// Launch
// ---------------------------------------------------------------------------
extern "C" void kernel_launch(void* const* d_in, const int* in_sizes, int n_in,
                              void* d_out, int out_size, void* d_ws, size_t ws_size,
                              hipStream_t stream) {
  const float* x      = (const float*)d_in[0];
  const float* W_tem1 = (const float*)d_in[1];   // (NHID, T)
  const float* W_tem2 = (const float*)d_in[2];   // (T, NHID)
  const float* W_cha1 = (const float*)d_in[3];   // (F, NHID)
  const float* W_cha2 = (const float*)d_in[4];   // (NHID, F)
  const float* W_tem3 = (const float*)d_in[5];   // (NHID, F)
  const float* b_tem3 = (const float*)d_in[6];   // (T,)
  const float* W_cha3 = (const float*)d_in[7];   // (NHID, T)
  const float* b_cha3 = (const float*)d_in[8];   // (F,)
  const float* W_t    = (const float*)d_in[9];   // (NHID, F)
  const float* G_t    = (const float*)d_in[10];  // (NHID, F)
  const float* gW_ih  = (const float*)d_in[11];  // (3*RNN, F+NHID)
  const float* gW_hh  = (const float*)d_in[12];  // (3*RNN, RNN)
  const float* gb_ih  = (const float*)d_in[13];  // (3*RNN,)
  const float* gb_hh  = (const float*)d_in[14];  // (3*RNN,)
  float* out = (float*)d_out;                    // (N, RNN)

  // Bump-allocate workspace (256B aligned); ~341 MB total.
  char* wp = (char*)d_ws;
  auto alloc = [&](size_t bytes) -> void* {
    void* r = (void*)wp;
    wp += (bytes + 255) & ~(size_t)255;
    return r;
  };
  _Float16* nx16    = (_Float16*)alloc((size_t)T * NF * 2);       // norm_x (t, c)
  _Float16* nxT16   = (_Float16*)alloc((size_t)NF * T * 2);       // norm_x (c, t)
  _Float16* x16     = (_Float16*)alloc((size_t)T * NF * 2);       // raw x f16
  _Float16* sb16    = (_Float16*)alloc((size_t)T * NF * 2);       // S_tem
  _Float16* si16    = (_Float16*)alloc((size_t)T * N * NHID * 2); // S_int
  _Float16* Atem16  = (_Float16*)alloc((size_t)T * T * 2);
  _Float16* BchaT16 = (_Float16*)alloc((size_t)F * F * 2);
  _Float16* Wt16    = (_Float16*)alloc((size_t)NHID * F * 2);
  _Float16* Gt16    = (_Float16*)alloc((size_t)NHID * F * 2);
  _Float16* Wih16   = (_Float16*)alloc((size_t)H3 * FC * 2);
  _Float16* Whh16   = (_Float16*)alloc((size_t)H3 * RNN * 2);
  float*    zt      = (float*)alloc((size_t)T * F * 4);
  float*    gtem    = (float*)alloc((size_t)T * 4);
  float*    gcha    = (float*)alloc((size_t)F * 4);
  float*    h0      = (float*)alloc((size_t)N * RNN * 4);
  float*    h1      = (float*)alloc((size_t)N * RNN * 4);
  _Float16* h16a    = (_Float16*)alloc((size_t)N * RNN * 2);
  _Float16* h16b    = (_Float16*)alloc((size_t)N * RNN * 2);

  // Normalization + f16 operand preparation.
  norm_kernel<<<(NF + 255) / 256, 256, 0, stream>>>(x, nx16, nxT16);
  cvt16_kernel<<<((size_t)T * NF + 255) / 256, 256, 0, stream>>>(x, x16, (size_t)T * NF);
  cvt16_kernel<<<(NHID * F + 255) / 256, 256, 0, stream>>>(W_t, Wt16, (size_t)NHID * F);
  cvt16_kernel<<<(NHID * F + 255) / 256, 256, 0, stream>>>(G_t, Gt16, (size_t)NHID * F);
  cvt16_kernel<<<(H3 * FC + 255) / 256, 256, 0, stream>>>(gW_ih, Wih16, (size_t)H3 * FC);
  cvt16_kernel<<<(H3 * RNN + 255) / 256, 256, 0, stream>>>(gW_hh, Whh16, (size_t)H3 * RNN);

  // Stats path.
  zt_kernel<<<T, F, 0, stream>>>(nx16, zt);
  gamma_kernel<<<1, 128, 0, stream>>>(zt, W_tem3, b_tem3, W_cha3, b_cha3, gtem, gcha);

  // Folded weight products.
  smallgemm16_kernel<<<(T * T + 255) / 256, 256, 0, stream>>>(W_tem2, W_tem1, Atem16, T, T, NHID, 0);
  smallgemm16_kernel<<<(F * F + 255) / 256, 256, 0, stream>>>(W_cha1, W_cha2, BchaT16, F, F, NHID, 1);

  // Fused WMMA GEMM passes.
  stem_kernel<<<NF / 16, 32, 0, stream>>>(nxT16, Atem16, gtem, sb16);
  fuse_kernel<<<(T * N) / 16, 32, 0, stream>>>(nx16, BchaT16, gcha, sb16, Wt16, Gt16, si16);

  // GRU: 64 sequential steps, ping-pong h (f32 + f16 shadow).
  fill0f_kernel<<<(N * RNN + 255) / 256, 256, 0, stream>>>(h0, N * RNN);
  fill0h_kernel<<<(N * RNN + 255) / 256, 256, 0, stream>>>(h16a, N * RNN);
  for (int t = 0; t < T; ++t) {
    float*    hin    = (t & 1) ? h1 : h0;
    _Float16* hin16  = (t & 1) ? h16b : h16a;
    float*    hout   = (t == T - 1) ? out : ((t & 1) ? h0 : h1);
    _Float16* hout16 = (t & 1) ? h16a : h16b;
    gru_step_kernel<<<dim3(N / 128, RNN / 16), 256, 0, stream>>>(
        t, x16, si16, Wih16, Whh16, gb_ih, gb_hh, hin, hin16, hout, hout16);
  }
}
